// Attention_20572893348773
// MI455X (gfx1250) — matrix operand order repaired and checked by reading, not compiled, not run on hardware
//
#include <hip/hip_runtime.h>
#include <hip/hip_bf16.h>
#include <math.h>

typedef __attribute__((ext_vector_type(16))) _Float16 v16h;
typedef __attribute__((ext_vector_type(8)))  _Float16 v8h;
typedef __attribute__((ext_vector_type(8)))  float    v8f;
typedef __attribute__((ext_vector_type(4)))  float    v4f;
typedef __attribute__((ext_vector_type(4)))  unsigned int u32x4;
typedef __attribute__((ext_vector_type(8)))  int      i32x8;
typedef __attribute__((ext_vector_type(4)))  int      i32x4;

#if defined(__has_builtin)
#if __has_builtin(__builtin_amdgcn_tensor_load_to_lds) && \
    __has_builtin(__builtin_amdgcn_s_wait_tensorcnt)
#define USE_TDM 1
#endif
#endif
#ifndef USE_TDM
#define USE_TDM 0
#endif

namespace {

constexpr int kB  = 2;
constexpr int kS  = 2048;
constexpr int kD  = 2048;
constexpr int kH  = 16;
constexpr int kHD = 128;
constexpr int kM  = kB * kS;         // 4096 rows of x
constexpr float kNegBig = -1.0e30f;

union AFrag { v16h v; v8h h[2]; };

__device__ __forceinline__ v8f wmma16x16x32(const AFrag& a, const AFrag& b, v8f c) {
  // D = A(16x32 f16) x B(32x16 f16) + C(16x16 f32)
  return __builtin_amdgcn_wmma_f32_16x16x32_f16(false, a.v, false, b.v,
                                                (short)0, c, false, false);
}

#if USE_TDM
// Issue one TDM 2-D tile load (f16 elements) global -> LDS with row padding.
// D# built per CDNA5 ISA ch.8: group0 = {count, lds_addr, global_addr, type=2},
// group1 = {data_size=1(2B), pad ctrl, tensor dims, tile dims, dim0 stride}.
// padIntervalCode: pad after 2<<code dwords; padAmountCode: pad (code+1) dwords.
__device__ __forceinline__ void tdm_load_tile_2d(
    unsigned int ldsByteAddr, const void* gptr,
    unsigned int tileW, unsigned int tileH, unsigned int strideElems,
    unsigned int padIntervalCode, unsigned int padAmountCode)
{
  const unsigned long long ga = (unsigned long long)(uintptr_t)gptr;
  u32x4 g0;
  g0[0] = 1u;                                      // count=1, user descriptor
  g0[1] = ldsByteAddr;                             // lds_addr (bytes)
  g0[2] = (unsigned int)ga;                        // global_addr[31:0]
  g0[3] = (unsigned int)((ga >> 32) & 0x01FFFFFFu) // global_addr[56:32]
        | (2u << 30);                              // type = 2 ("image")
  i32x8 g1;
  const unsigned int tw = 0x7FFFFFFFu;             // huge tensor dims: no OOB
  g1[0] = (int)((1u << 16)                         // data_size = 1 -> 2 bytes
        | (1u << 20)                               // pad_enable
        | (padIntervalCode << 22)
        | (padAmountCode << 25));
  g1[1] = (int)((tw & 0xFFFFu) << 16);             // tensor_dim0[15:0]
  g1[2] = (int)((tw >> 16) | ((tw & 0xFFFFu) << 16)); // dim0 hi | dim1 lo
  g1[3] = (int)((tw >> 16) | (tileW << 16));       // dim1 hi | tile_dim0
  g1[4] = (int)(tileH & 0xFFFFu);                  // tile_dim1 (tile_dim2 = 0)
  g1[5] = (int)strideElems;                        // tensor_dim0_stride[31:0]
  g1[6] = 0;
  g1[7] = 0;
  const i32x4 gz = {0, 0, 0, 0};
#if __clang_major__ >= 23
  const i32x8 gz8 = {0, 0, 0, 0, 0, 0, 0, 0};
  __builtin_amdgcn_tensor_load_to_lds(g0, g1, gz, gz, gz8, 0);
#else
  __builtin_amdgcn_tensor_load_to_lds(g0, g1, gz, gz, 0);
#endif
}
#endif

// ---------------------------------------------------------------------------
// GEMM:  out[m,n] = sum_k A[m,k] * W[n,k]      (y = A @ W^T)
// A_F16: 0 -> A is f32 (x), 1 -> A is f16 (attention output; staged via TDM)
// MODE : 0 -> store f16 to qkv workspace [B,H,S,HD]
//        1 -> same as 0 but apply RoPE first (for Q and K)
//        2 -> store f32 to d_out (final projection)
// ---------------------------------------------------------------------------
template <int A_F16, int MODE>
__global__ __launch_bounds__(256)
void gemm_x_wt(const float* __restrict__ Af32, const _Float16* __restrict__ Ah16,
               const float* __restrict__ W, _Float16* __restrict__ ws_out,
               float* __restrict__ f32_out, const float* __restrict__ cosT,
               const float* __restrict__ sinT)
{
  __shared__ _Float16 At[128][40];    // A tile (f16): 64B rows + 16B pad
  __shared__ _Float16 Bt[32][136];    // W tile stored transposed: Bt[k][n]

  const int tid  = threadIdx.x;
  const int wv_  = tid >> 5;          // wave id 0..7
  const int lane = tid & 31;
  const int M0 = blockIdx.x * 128;
  const int N0 = blockIdx.y * 128;
  const int wm = (wv_ >> 1) * 32;     // wave row offset within tile
  const int wn = (wv_ & 1) * 64;      // wave col offset within tile

  v8f acc[2][4];
#pragma unroll
  for (int i = 0; i < 2; ++i)
#pragma unroll
    for (int j = 0; j < 4; ++j)
      acc[i][j] = v8f{0.f,0.f,0.f,0.f,0.f,0.f,0.f,0.f};

  const int arow = tid >> 1;          // 0..127
  const int acol = (tid & 1) * 16;    // 0 or 16

  for (int k0 = 0; k0 < kD; k0 += 32) {
    // ---- stage A tile (128x32) into LDS as f16 ----
    if constexpr (A_F16) {
#if USE_TDM
      if (wv_ == 0) {
        // 128 rows x 32 f16; row = 64B = 16 dwords -> interval code 3,
        // pad 4 dwords (code 3) -> LDS row stride 40 halfs == At[][40].
        tdm_load_tile_2d((unsigned int)(uintptr_t)&At[0][0],
                         Ah16 + (size_t)M0 * kD + k0,
                         32u, 128u, (unsigned int)kD, 3u, 3u);
      }
#else
      {
        const _Float16* src = Ah16 + (size_t)(M0 + arow) * kD + k0 + acol;
        v8h a0 = *(const v8h*)src;
        v8h a1 = *(const v8h*)(src + 8);
        *(v8h*)&At[arow][acol]     = a0;
        *(v8h*)&At[arow][acol + 8] = a1;
      }
#endif
    } else {
      const float* src = Af32 + (size_t)(M0 + arow) * kD + k0 + acol;
      _Float16 ah[16];
#pragma unroll
      for (int j = 0; j < 16; j += 4) {
        v4f f = *(const v4f*)(src + j);
        ah[j]     = (_Float16)f.x; ah[j + 1] = (_Float16)f.y;
        ah[j + 2] = (_Float16)f.z; ah[j + 3] = (_Float16)f.w;
      }
      v8h t0, t1;
#pragma unroll
      for (int j = 0; j < 8; ++j) { t0[j] = ah[j]; t1[j] = ah[8 + j]; }
      *(v8h*)&At[arow][acol]     = t0;
      *(v8h*)&At[arow][acol + 8] = t1;
    }
    // ---- stage W tile transposed: Bt[k][n] = W[N0+n][k0+k] (f32->f16) ----
    {
      const float* src = W + (size_t)(N0 + arow) * kD + k0 + acol;
#pragma unroll
      for (int j = 0; j < 16; j += 4) {
        v4f f = *(const v4f*)(src + j);
        Bt[acol + j    ][arow] = (_Float16)f.x;
        Bt[acol + j + 1][arow] = (_Float16)f.y;
        Bt[acol + j + 2][arow] = (_Float16)f.z;
        Bt[acol + j + 3][arow] = (_Float16)f.w;
      }
    }
#if USE_TDM
    if constexpr (A_F16) {
      if (wv_ == 0) __builtin_amdgcn_s_wait_tensorcnt(0);
    }
#endif
    __syncthreads();

    // ---- fragments + 8 WMMAs for this K slab ----
    AFrag a[2], b[4];
    const int mloc = wm + (lane & 15);
    const int ko   = (lane >> 4) * 8;     // K offset per lane-half (ISA layout)
#pragma unroll
    for (int i = 0; i < 2; ++i) {
      a[i].h[0] = *(const v8h*)&At[mloc + i * 16][ko];        // K ko..ko+7
      a[i].h[1] = *(const v8h*)&At[mloc + i * 16][ko + 16];   // K ko+16..ko+23
    }
#pragma unroll
    for (int j = 0; j < 4; ++j) {
      const int nloc = wn + j * 16 + (lane & 15);
      for (int s = 0; s < 8; ++s) {
        b[j].h[0][s] = Bt[ko + s][nloc];
        b[j].h[1][s] = Bt[ko + 16 + s][nloc];
      }
    }
#pragma unroll
    for (int i = 0; i < 2; ++i)
#pragma unroll
      for (int j = 0; j < 4; ++j)
        acc[i][j] = wmma16x16x32(a[i], b[j], acc[i][j]);
    __syncthreads();
  }

  // ---- epilogue ----
#pragma unroll
  for (int i = 0; i < 2; ++i) {
#pragma unroll
    for (int j = 0; j < 4; ++j) {
      const int col   = N0 + wn + j * 16 + (lane & 15);       // C/D: lane = N
      const int rbase = M0 + wm + i * 16 + (lane >> 4) * 8;   // VGPR r = M
#pragma unroll
      for (int r = 0; r < 8; ++r) {
        const int row = rbase + r;
        float val = acc[i][j][r];
        if constexpr (MODE == 2) {
          f32_out[(size_t)row * kD + col] = val;
        } else {
          if constexpr (MODE == 1) {
            // RoPE: pairs (even,odd) of head-dim live in adjacent lanes
            const int s = row & (kS - 1);
            const int p = (col & (kHD - 1)) >> 1;
            const float c  = cosT[s * (kHD / 2) + p];
            const float sn = sinT[s * (kHD / 2) + p];
            const float other = __shfl_xor(val, 1);
            val = (col & 1) ? (other * sn + val * c)   // odd:  re*sn + im*c
                            : (val * c - other * sn);  // even: re*c  - im*sn
          }
          const int bb = row >> 11;          // row / S
          const int s  = row & (kS - 1);
          const int h  = col >> 7;           // col / HD
          const int hd = col & (kHD - 1);
          ws_out[(((size_t)(bb * kH + h)) * kS + s) * kHD + hd] = (_Float16)val;
        }
      }
    }
  }
}

// ---------------------------------------------------------------------------
// Flash attention over one (b,h): Q,K,V in [B*H, S, HD] f16.
// Workgroup = 128 query rows; wave w owns rows m0..m0+15, keeps Q fragments
// in registers, streams 32-key K/V tiles through LDS with online softmax.
// V tiles are moved by the Tensor Data Mover (padded-LDS descriptor).
// ---------------------------------------------------------------------------
__global__ __launch_bounds__(256)
void flash_attn(const _Float16* __restrict__ Q, const _Float16* __restrict__ K,
                const _Float16* __restrict__ V, _Float16* __restrict__ AO)
{
  __shared__ _Float16 Kt[128][40];     // K tile transposed: Kt[hd][t]
  __shared__ _Float16 Vt[32][136];     // V tile row-major: 256B rows + 16B pad
  __shared__ _Float16 Pl[8][16][40];   // per-wave P relayout scratch

  const int tid  = threadIdx.x;
  const int wv_  = tid >> 5;
  const int lane = tid & 31;
  const int M0 = blockIdx.x * 128;
  const int bh = blockIdx.y;
  const size_t base = (size_t)bh * kS * kHD;
  const int m0 = M0 + wv_ * 16;

  // Q A-fragments: 4 chunks of K=32 over HD=128, loaded once.
  AFrag qf[4];
  {
    const int m  = m0 + (lane & 15);
    const int ko = (lane >> 4) * 8;
    const _Float16* qp = Q + base + (size_t)m * kHD;
#pragma unroll
    for (int kc = 0; kc < 4; ++kc) {
      qf[kc].h[0] = *(const v8h*)(qp + kc * 32 + ko);
      qf[kc].h[1] = *(const v8h*)(qp + kc * 32 + ko + 16);
    }
  }

  v8f oacc[8];
#pragma unroll
  for (int j = 0; j < 8; ++j) oacc[j] = v8f{0.f,0.f,0.f,0.f,0.f,0.f,0.f,0.f};
  float mx[8], l[8];
#pragma unroll
  for (int r = 0; r < 8; ++r) { mx[r] = -3.0e38f; l[r] = 0.f; }
  const float scale = 0.088388347648318447f;   // 1/sqrt(128)

  const int nt = M0 / 32 + 4;                  // causal: keys <= M0+127
  for (int it = 0; it < nt; ++it) {
    const int t0 = it * 32;
#if USE_TDM
    if (wv_ == 0) {
      // 32 rows x 128 f16; row = 256B = 64 dwords -> interval code 5,
      // pad 4 dwords (code 3) -> LDS row stride 136 halfs == Vt[][136].
      tdm_load_tile_2d((unsigned int)(uintptr_t)&Vt[0][0],
                       V + base + (size_t)t0 * kHD,
                       (unsigned int)kHD, 32u, (unsigned int)kHD, 5u, 3u);
    }
#endif
    // ---- stage K transposed (TDM cannot transpose); V manual if no TDM ----
    {
      const int tl = tid >> 3;                 // key 0..31
      const int hg = (tid & 7) * 16;           // hd group
      const _Float16* kp = K + base + (size_t)(t0 + tl) * kHD + hg;
      v8h ka = *(const v8h*)kp;
      v8h kb = *(const v8h*)(kp + 8);
#pragma unroll
      for (int j = 0; j < 8; ++j) {
        Kt[hg + j][tl]     = ka[j];
        Kt[hg + 8 + j][tl] = kb[j];
      }
#if !USE_TDM
      const _Float16* vp = V + base + (size_t)(t0 + tl) * kHD + hg;
      *(v8h*)&Vt[tl][hg]     = *(const v8h*)vp;
      *(v8h*)&Vt[tl][hg + 8] = *(const v8h*)(vp + 8);
#endif
    }
#if USE_TDM
    if (wv_ == 0) __builtin_amdgcn_s_wait_tensorcnt(0);
#endif
    __syncthreads();

    // ---- scores: two 16x16 f32 tiles (keys t0..t0+15, t0+16..t0+31) ----
    v8f c0 = v8f{0.f,0.f,0.f,0.f,0.f,0.f,0.f,0.f};
    v8f c1 = v8f{0.f,0.f,0.f,0.f,0.f,0.f,0.f,0.f};
#pragma unroll
    for (int kc = 0; kc < 4; ++kc) {
      AFrag b0, b1;
      const int hd = kc * 32 + (lane >> 4) * 8;
      const int tn = lane & 15;
      for (int s = 0; s < 8; ++s) {
        b0.h[0][s] = Kt[hd + s][tn];
        b0.h[1][s] = Kt[hd + 16 + s][tn];
        b1.h[0][s] = Kt[hd + s][tn + 16];
        b1.h[1][s] = Kt[hd + 16 + s][tn + 16];
      }
      c0 = wmma16x16x32(qf[kc], b0, c0);
      c1 = wmma16x16x32(qf[kc], b1, c1);
    }

    // ---- causal mask + online softmax ----
    const int tcol = t0 + (lane & 15);
    const int mrb  = m0 + (lane >> 4) * 8;
    float p0[8], p1[8];
#pragma unroll
    for (int r = 0; r < 8; ++r) {
      const int m = mrb + r;
      float s0 = c0[r] * scale; if (tcol > m)      s0 = kNegBig;
      float s1 = c1[r] * scale; if (tcol + 16 > m) s1 = kNegBig;
      float rowm = fmaxf(s0, s1);
#pragma unroll
      for (int off = 1; off < 16; off <<= 1)
        rowm = fmaxf(rowm, __shfl_xor(rowm, off));
      const float newm  = fmaxf(mx[r], rowm);
      const float alpha = __expf(mx[r] - newm);
      mx[r] = newm;
      p0[r] = __expf(s0 - newm);
      p1[r] = __expf(s1 - newm);
      float rs = p0[r] + p1[r];
#pragma unroll
      for (int off = 1; off < 16; off <<= 1)
        rs += __shfl_xor(rs, off);
      l[r] = l[r] * alpha + rs;
#pragma unroll
      for (int j = 0; j < 8; ++j) oacc[j][r] *= alpha;
    }

    // ---- relayout P (C-layout) -> A fragment through per-wave LDS ----
#pragma unroll
    for (int r = 0; r < 8; ++r) {
      const int mr = (lane >> 4) * 8 + r;
      Pl[wv_][mr][lane & 15]        = (_Float16)p0[r];
      Pl[wv_][mr][16 + (lane & 15)] = (_Float16)p1[r];
    }
    asm volatile("s_wait_dscnt 0" ::: "memory");  // wave-internal DS RAW

    AFrag pf;
    {
      const int m  = lane & 15;
      const int ko = (lane >> 4) * 8;
      pf.h[0] = *(const v8h*)&Pl[wv_][m][ko];
      pf.h[1] = *(const v8h*)&Pl[wv_][m][ko + 16];
    }
    // ---- O += P(16x32) @ V(32x128): 8 WMMAs, V needs no transpose ----
#pragma unroll
    for (int j = 0; j < 8; ++j) {
      AFrag vf;
      for (int s = 0; s < 8; ++s) {
        vf.h[0][s] = Vt[(lane >> 4) * 8 + s][j * 16 + (lane & 15)];
        vf.h[1][s] = Vt[(lane >> 4) * 8 + 16 + s][j * 16 + (lane & 15)];
      }
      oacc[j] = wmma16x16x32(pf, vf, oacc[j]);
    }
    __syncthreads();
  }

  // ---- normalize and store to [ (b,s), (h,hd) ] f16 for final GEMM ----
  const int bb = bh / kH;
  const int h  = bh % kH;
  float inv[8];
#pragma unroll
  for (int r = 0; r < 8; ++r) inv[r] = 1.0f / l[r];
#pragma unroll
  for (int j = 0; j < 8; ++j) {
    const int hd = j * 16 + (lane & 15);
#pragma unroll
    for (int r = 0; r < 8; ++r) {
      const int s = m0 + (lane >> 4) * 8 + r;
      AO[((size_t)(bb * kS + s)) * kD + h * kHD + hd] =
          (_Float16)(oacc[j][r] * inv[r]);
    }
  }
}

} // namespace

extern "C" void kernel_launch(void* const* d_in, const int* in_sizes, int n_in,
                              void* d_out, int out_size, void* d_ws, size_t ws_size,
                              hipStream_t stream)
{
  (void)in_sizes; (void)n_in; (void)out_size; (void)ws_size;
  const float* x    = (const float*)d_in[0];
  const float* fcos = (const float*)d_in[1];
  const float* fsin = (const float*)d_in[2];
  // d_in[3] = mask: unused, causal mask applied analytically
  const float* wq   = (const float*)d_in[4];
  const float* wk   = (const float*)d_in[5];
  const float* wv   = (const float*)d_in[6];
  const float* wo   = (const float*)d_in[7];
  // d_in[8] = start_pos (always 0 here)

  const size_t qkvElems = (size_t)kB * kH * kS * kHD;   // 8.39M f16 each
  _Float16* qws  = (_Float16*)d_ws;
  _Float16* kws  = qws + qkvElems;
  _Float16* vws  = kws + qkvElems;
  _Float16* aows = vws + qkvElems;                       // [M, D] f16
  float* out = (float*)d_out;

  dim3 blk(256);
  dim3 g(kM / 128, kD / 128);       // 32 x 16
  // Q,K projections with fused RoPE; V plain.
  gemm_x_wt<0, 1><<<g, blk, 0, stream>>>(x, nullptr, wq, qws, nullptr, fcos, fsin);
  gemm_x_wt<0, 1><<<g, blk, 0, stream>>>(x, nullptr, wk, kws, nullptr, fcos, fsin);
  gemm_x_wt<0, 0><<<g, blk, 0, stream>>>(x, nullptr, wv, vws, nullptr, nullptr, nullptr);

  dim3 ga(kS / 128, kB * kH);       // 16 x 32
  flash_attn<<<ga, blk, 0, stream>>>(qws, kws, vws, aows);

  gemm_x_wt<1, 2><<<g, blk, 0, stream>>>(nullptr, aows, wo, nullptr, out, nullptr, nullptr);
}
